// SFFNet_42649025250128
// MI455X (gfx1250) — matrix operand order repaired and checked by reading, not compiled, not run on hardware
//
#include <hip/hip_runtime.h>
#include <hip/hip_bf16.h>

// ---------------------------------------------------------------------------
// CDNA5 (gfx1250) implementation of the SFFNet block.
// All GEMM-like stages run through v_wmma_f32_16x16x32_bf16 (wave32).
// Fragment loads are ds_load_b128-shaped; B-panels are K-transposed in LDS.
// Pure-copy panel fills use GLOBAL_LOAD_ASYNC_TO_LDS_B128 when available.
// ---------------------------------------------------------------------------

typedef __bf16 bf16;
typedef __attribute__((ext_vector_type(16))) __bf16 bf16x16;
typedef __attribute__((ext_vector_type(8)))  __bf16 bf16x8;
typedef __attribute__((ext_vector_type(4)))  __bf16 bf16x4;
typedef __attribute__((ext_vector_type(8)))  float  f32x8;
typedef __attribute__((ext_vector_type(4)))  unsigned int u32x4;
typedef __attribute__((ext_vector_type(4)))  int i32x4;

#if defined(__AMDGCN__) &&                                             \
    __has_builtin(__builtin_amdgcn_global_load_async_to_lds_b128) &&   \
    __has_builtin(__builtin_amdgcn_s_wait_asynccnt)
#define HAVE_ASYNC 1
#endif

#ifdef HAVE_ASYNC
static __device__ inline void async_cp16(const void* g, void* l) {
  // Builtin signature (from hipcc diagnostic): (v4i*, v4i*, Ii offset, Ii cpol)
  __builtin_amdgcn_global_load_async_to_lds_b128((i32x4*)g, (i32x4*)l, 0, 0);
}
static __device__ inline void async_wait() {
  __builtin_amdgcn_s_wait_asynccnt(0);
}
#else
static __device__ inline void async_cp16(const void* g, void* l) {
  *(u32x4*)l = *(const u32x4*)g;
}
static __device__ inline void async_wait() {}
#endif

#define LANE (threadIdx.x & 31)

static __device__ inline f32x8 wmma_bf16(bf16x16 a, bf16x16 b, f32x8 c) {
  // (neg_a, A, neg_b, B, c_mod, C, reuse_a, reuse_b)
  return __builtin_amdgcn_wmma_f32_16x16x32_bf16(false, a, false, b, (short)0, c,
                                                 false, false);
}

static __device__ inline bf16x16 cat8(bf16x8 lo, bf16x8 hi) {
  return __builtin_shufflevector(lo, hi, 0, 1, 2, 3, 4, 5, 6, 7, 8, 9, 10, 11,
                                 12, 13, 14, 15);
}

// Lane-xor exchange via ds_swizzle (group-of-32 xor-mask mode): no index
// setup, no bounds-check selects.  new_lane = (lane & 31) ^ M.
template <int M>
static __device__ inline float xshuf(float v) {
  return __int_as_float(
      __builtin_amdgcn_ds_swizzle(__float_as_int(v), (M << 10) | 31));
}

// A-fragment, 16x32 bf16 slice of a row-major [16][ld] panel (ld multiple of 8).
// Contiguous-K order used here: lane l -> row (l&15); K = (l>>4)*16 + half.
//             Two b128 LDS loads.
static __device__ inline bf16x16 frag_a(const bf16* A, int ld) {
  const bf16* p = A + (LANE & 15) * ld + ((LANE >> 4) * 16);
  return cat8(*(const bf16x8*)p, *(const bf16x8*)(p + 8));
}

// B-fragment from a K-TRANSPOSED panel Bt[n][32] (row n = output column).
// Contiguous-K order (same as frag_a): lane l -> col (l&15); K = (l>>4)*16 + half.
// Lane reads 16 contiguous bf16 -> two b128 LDS loads.
static __device__ inline bf16x16 frag_bt(const bf16* Bt, int ncol) {
  const bf16* p = Bt + (ncol + (LANE & 15)) * 32 + (LANE >> 4) * 16;
  return cat8(*(const bf16x8*)p, *(const bf16x8*)(p + 8));
}

// ---------------------------------------------------------------------------
// prep kernels
// ---------------------------------------------------------------------------
__global__ void k_f32_to_bf16(const float* __restrict__ in,
                              bf16* __restrict__ out, long n4) {
  long i = (long)blockIdx.x * blockDim.x + threadIdx.x;
  if (i < n4) {
    float4 f = ((const float4*)in)[i];
    bf16x4 o;
    o[0] = (bf16)f.x; o[1] = (bf16)f.y; o[2] = (bf16)f.z; o[3] = (bf16)f.w;
    ((bf16x4*)out)[i] = o;
  }
}

// bias[h][i][j] = table[relIdx(i,j)*16 + h], Swin relative-position index.
__global__ void k_bias_prep(const float* __restrict__ tbl,
                            float* __restrict__ biasTab) {
  int i = blockIdx.x * 256 + threadIdx.x;   // 16*64*64 = 65536
  int h = i >> 12;
  int ii = (i >> 6) & 63;
  int jj = i & 63;
  int di = (ii >> 3) - (jj >> 3) + 7;
  int dj = (ii & 7) - (jj & 7) + 7;
  biasTab[i] = tbl[(di * 15 + dj) * 16 + h];
}

// ---------------------------------------------------------------------------
// Generic NCHW GEMM: Y[b][m][hw] = sum_k W[m][k] * X[b][k][hw]
// Tile 128 (M) x 64 (N). 8 waves; wave w owns a 32x32 region (2x2 WMMA tiles).
// ---------------------------------------------------------------------------
template <bool OUT_F32>
__global__ void __launch_bounds__(256)
k_gemm_nchw(const bf16* __restrict__ W, const bf16* __restrict__ X,
            bf16* __restrict__ Yb, float* __restrict__ Yf,
            int M, int K, int HW) {
  __shared__ __align__(16) bf16 sA[128 * 32];
  __shared__ __align__(16) bf16 sBt[64 * 32];    // K-transposed: [n][k]
  const int tid = threadIdx.x;
  const int wav = tid >> 5;
  const int m0 = blockIdx.y * 128;
  const long n0 = (long)blockIdx.x * 64;
  const int b = (int)(n0 / HW);        // 64 | HW, so a tile never crosses b
  const int rem0 = (int)(n0 % HW);
  const bf16* Xb = X + (long)b * K * HW;

  const int mB = (wav & 3) * 32;
  const int nB = (wav >> 2) * 32;
  f32x8 acc00 = {}, acc01 = {}, acc10 = {}, acc11 = {};

  for (int k0 = 0; k0 < K; k0 += 32) {
    __syncthreads();
    // A panel: 128x32 = 512 16B-chunks, 2 per thread (async b128 copies)
#pragma unroll
    for (int c = tid; c < 512; c += 256) {
      int r = c >> 2, ko = (c & 3) << 3;
      async_cp16(&W[(long)(m0 + r) * K + k0 + ko], &sA[r * 32 + ko]);
    }
    // B panel: coalesced b128 global load, transpose into sBt[n][k]
    {
      int kk = tid >> 3, j = (tid & 7) << 3;
      bf16x8 vv = *(const bf16x8*)&Xb[(long)(k0 + kk) * HW + rem0 + j];
#pragma unroll
      for (int q = 0; q < 8; ++q) sBt[(j + q) * 32 + kk] = vv[q];
    }
    __builtin_prefetch(&Xb[(long)(k0 + 32 + (tid >> 3)) * HW + rem0], 0, 0);
    async_wait();
    __syncthreads();
    bf16x16 a0 = frag_a(sA + (mB + 0) * 32, 32);
    bf16x16 a1 = frag_a(sA + (mB + 16) * 32, 32);
    bf16x16 b0 = frag_bt(sBt, nB);
    bf16x16 b1 = frag_bt(sBt, nB + 16);
    acc00 = wmma_bf16(a0, b0, acc00);
    acc01 = wmma_bf16(a0, b1, acc01);
    acc10 = wmma_bf16(a1, b0, acc10);
    acc11 = wmma_bf16(a1, b1, acc11);
  }

  const int rowOff = (LANE >> 4) * 8;
  const int col = LANE & 15;
  f32x8 accs[4] = {acc00, acc01, acc10, acc11};
#pragma unroll
  for (int s = 0; s < 4; ++s) {
    int sm = s >> 1, sn = s & 1;
    int mBase = m0 + mB + sm * 16 + rowOff;
    int nIdx = rem0 + nB + sn * 16 + col;
#pragma unroll
    for (int v = 0; v < 8; ++v) {
      long o = (long)(b * M + mBase + v) * HW + nIdx;
      if (OUT_F32) Yf[o] = accs[s][v];
      else         Yb[o] = (bf16)accs[s][v];
    }
  }
}

// ---------------------------------------------------------------------------
// Window attention: one block per (window, head); 4 waves, wave = 16-row tile.
// Q,K stored [tok][32] with dd 16..31 zero-filled (free K=16->32 padding);
// V stored [dd][tok].  All fragment loads are b128-shaped, no predication.
// ---------------------------------------------------------------------------
__global__ void __launch_bounds__(128)
k_attn(const bf16* __restrict__ QKV, const float* __restrict__ biasTab,
       bf16* __restrict__ O) {
  const int C = 256, Wd = 256, HW = 256 * 256;
  __shared__ __align__(16) bf16 sQt[64][32];     // [tok][dd], dd>=16 zero
  __shared__ __align__(16) bf16 sKt[64][32];     // [tok][dd], dd>=16 zero
  __shared__ __align__(16) bf16 sV[16][64];      // [dd][tok]
  __shared__ __align__(16) bf16 sP[4][16][64];   // per-wave P tiles
  __shared__ __align__(16) bf16 sO[64][16];      // [tok][dd] output staging
  const int lane = LANE, wav = threadIdx.x >> 5;
  const int tid = threadIdx.x;
  const int h = blockIdx.x & 15;
  const int win = blockIdx.x >> 4;
  const int b = win >> 10;
  const int wr = (win >> 5) & 31;
  const int wc = win & 31;
  const int mt = wav;
  const long qkvBase = (long)(b * 768 + (h << 4)) * HW +
                       (long)(wr << 3) * Wd + (wc << 3);

  // Stage V natural layout: 128 16B chunks, 1/thread (async b128 copies).
  {
    int dd = tid >> 3, trow = tid & 7;
    async_cp16(&QKV[qkvBase + (long)(512 + dd) * HW + trow * Wd],
               &sV[dd][trow * 8]);
  }
  // Zero dd 16..31 of sQt/sKt: 128 rows-of-16, one per thread (2 b128 stores).
  {
    int sel = tid >> 6, row = tid & 63;
    bf16* dst = (sel ? &sKt[row][16] : &sQt[row][16]);
    u32x4 z = {};
    *(u32x4*)dst = z;
    *(u32x4*)(dst + 8) = z;
  }
  // Stage Q,K transposed: 2 sel x 16 dd x 8 rows = 256 16B chunks, 2/thread.
#pragma unroll
  for (int e = tid; e < 256; e += 128) {
    int sel = e >> 7, dd = (e >> 3) & 15, trow = e & 7;
    bf16x8 vv = *(const bf16x8*)&QKV[qkvBase + (long)(sel * 256 + dd) * HW +
                                     trow * Wd];
    bf16(*dst)[32] = sel ? sKt : sQt;
#pragma unroll
    for (int q = 0; q < 8; ++q) dst[trow * 8 + q][dd] = vv[q];
  }
  async_wait();
  __syncthreads();

  // dots = Q K^T (K contraction 16, zero-padded to 32)
  bf16x16 qa = frag_a(&sQt[mt * 16][0], 32);
  const float scale = 0.25f;  // d^-0.5, d = 16
  f32x8 d[4];
#pragma unroll
  for (int t = 0; t < 4; ++t) {
    f32x8 z = {};
    d[t] = wmma_bf16(qa, frag_bt(&sKt[0][0], t * 16), z);
  }

  const int rowOff = (lane >> 4) * 8, colOff = lane & 15;
#pragma unroll
  for (int t = 0; t < 4; ++t)
#pragma unroll
    for (int v = 0; v < 8; ++v) {
      int i = mt * 16 + rowOff + v;
      int j = t * 16 + colOff;
      d[t][v] = d[t][v] * scale + biasTab[(h << 12) + (i << 6) + j];
    }

  // Row softmax: each row lives in one 16-lane half; xor-swizzle reductions.
#pragma unroll
  for (int v = 0; v < 8; ++v) {
    float mx = -3.4e38f;
#pragma unroll
    for (int t = 0; t < 4; ++t) mx = fmaxf(mx, d[t][v]);
    mx = fmaxf(mx, xshuf<1>(mx));
    mx = fmaxf(mx, xshuf<2>(mx));
    mx = fmaxf(mx, xshuf<4>(mx));
    mx = fmaxf(mx, xshuf<8>(mx));
    float sum = 0.f;
#pragma unroll
    for (int t = 0; t < 4; ++t) {
      float e = __expf(d[t][v] - mx);
      d[t][v] = e;
      sum += e;
    }
    sum += xshuf<1>(sum);
    sum += xshuf<2>(sum);
    sum += xshuf<4>(sum);
    sum += xshuf<8>(sum);
    float inv = 1.0f / sum;
#pragma unroll
    for (int t = 0; t < 4; ++t) d[t][v] *= inv;
  }

  // P (C-layout) -> LDS (row-major) -> A-layout fragments.
#pragma unroll
  for (int t = 0; t < 4; ++t)
#pragma unroll
    for (int v = 0; v < 8; ++v)
      sP[wav][rowOff + v][t * 16 + colOff] = (bf16)d[t][v];
  __syncthreads();

  // O = P @ V, K = 64 as two K=32 slices.
  f32x8 o = {};
#pragma unroll
  for (int c = 0; c < 2; ++c) {
    bf16x16 pa = frag_a(&sP[wav][0][0] + c * 32, 64);
    // V B-fragment: lane col = dd, K = tok; 16 contiguous toks in sV[dd].
    const bf16* p = &sV[lane & 15][c * 32 + (lane >> 4) * 16];
    bf16x16 vb = cat8(*(const bf16x8*)p, *(const bf16x8*)(p + 8));
    o = wmma_bf16(pa, vb, o);
  }

  // Stage O tile, then write coalesced 16B rows.
#pragma unroll
  for (int v = 0; v < 8; ++v) sO[mt * 16 + rowOff + v][colOff] = (bf16)o[v];
  __syncthreads();
  {
    int cc = tid >> 3, trow = tid & 7;
    bf16x8 vv;
#pragma unroll
    for (int q = 0; q < 8; ++q) vv[q] = sO[trow * 8 + q][cc];
    *(bf16x8*)&O[(long)(b * C + (h << 4) + cc) * HW + ((wr << 3) + trow) * Wd +
                 (wc << 3)] = vv;
  }
}

// ---------------------------------------------------------------------------
// Fused strip convolutions as implicit GEMM (K = C*8 = 2048 each), wx
// (vertical, reflect row at H) then wy (horizontal, reflect col at W) feeding
// the same accumulators. Tile: 64 cout x 64 contiguous x of one row y.
// ---------------------------------------------------------------------------
__global__ void __launch_bounds__(256)
k_strip(const bf16* __restrict__ Wx, const bf16* __restrict__ Wy,
        const bf16* __restrict__ O, const float* __restrict__ bx,
        const float* __restrict__ by, bf16* __restrict__ Out) {
  const int C = 256, H = 256, Wd = 256, HW = H * Wd;
  __shared__ __align__(16) bf16 sA[64 * 32];
  __shared__ __align__(16) bf16 sBt[64 * 32];    // K-transposed: [n][k]
  __shared__ __align__(16) bf16 sRow[4][72];     // horizontal row windows
  const int tid = threadIdx.x, wav = tid >> 5;
  const int m0 = blockIdx.y * 64;
  const long s0 = (long)blockIdx.x * 64;
  const int b = (int)(s0 / HW);
  const int rem = (int)(s0 % HW);
  const int y = rem >> 8;
  const int x0 = rem & 255;
  const bf16* Ob = O + (long)b * C * HW;

  const int mB = (wav & 3) * 16;
  const int nB = (wav >> 2) * 32;
  f32x8 acc0 = {}, acc1 = {};

  // ---- pass 0: vertical strip conv (wx) ----
  for (int k0 = 0; k0 < 2048; k0 += 32) {
    __syncthreads();
    {
      int r = tid >> 2, ko = (tid & 3) << 3;
      async_cp16(&Wx[(long)(m0 + r) * 2048 + k0 + ko], &sA[r * 32 + ko]);
    }
    // B: one b128 global load per thread (row-validity uniform per chunk),
    // transposed into sBt.
    {
      int kk = tid >> 3, j0 = (tid & 7) << 3;
      int ci = (k0 + kk) >> 3, rr = kk & 7;
      int t = y + rr - 3;
      bf16x8 vv;
#pragma unroll
      for (int q = 0; q < 8; ++q) vv[q] = (bf16)0.f;
      if (t >= 0 && t <= H) {
        int tr = (t == H) ? H - 2 : t;
        vv = *(const bf16x8*)&Ob[(long)ci * HW + tr * Wd + x0 + j0];
      }
#pragma unroll
      for (int q = 0; q < 8; ++q) sBt[(j0 + q) * 32 + kk] = vv[q];
    }
    __builtin_prefetch(&Wx[(long)(m0 + (tid >> 2)) * 2048 + k0 + 32], 0, 0);
    async_wait();
    __syncthreads();
    bf16x16 a = frag_a(sA + mB * 32, 32);
    acc0 = wmma_bf16(a, frag_bt(sBt, nB), acc0);
    acc1 = wmma_bf16(a, frag_bt(sBt, nB + 16), acc1);
  }

  // ---- pass 1: horizontal strip conv (wy) ----
  for (int k0 = 0; k0 < 2048; k0 += 32) {
    __syncthreads();
    {
      int r = tid >> 2, ko = (tid & 3) << 3;
      async_cp16(&Wy[(long)(m0 + r) * 2048 + k0 + ko], &sA[r * 32 + ko]);
    }
    // Stage the 4 channel-rows [x0-3 .. x0+68] with reflect at W.
#pragma unroll
    for (int e = tid; e < 4 * 72; e += 256) {
      int cl = e / 72, xo = e % 72;
      int ci = (k0 >> 3) + cl;
      int u = x0 + xo - 3;
      bf16 v = (bf16)0.f;
      if (u >= 0 && u <= Wd) {
        int uc = (u == Wd) ? Wd - 2 : u;
        v = Ob[(long)ci * HW + y * Wd + uc];
      }
      sRow[cl][xo] = v;
    }
    async_wait();
    __syncthreads();
    bf16x16 a = frag_a(sA + mB * 32, 32);
    bf16x16 b0, b1;
#pragma unroll
    for (int i = 0; i < 16; ++i) {
      int kk = (LANE >> 4) * 16 + i;
      int cl = kk >> 3, rr = kk & 7;
      b0[i] = sRow[cl][nB + (LANE & 15) + rr];
      b1[i] = sRow[cl][nB + 16 + (LANE & 15) + rr];
    }
    acc0 = wmma_bf16(a, b0, acc0);
    acc1 = wmma_bf16(a, b1, acc1);
  }

  const int rowOff = (LANE >> 4) * 8, col = LANE & 15;
#pragma unroll
  for (int v = 0; v < 8; ++v) {
    int m = m0 + mB + rowOff + v;
    float bias = bx[m] + by[m];
    int xa = x0 + nB + col;
    long base = (long)(b * C + m) * HW + y * Wd;
    Out[base + xa] = (bf16)(acc0[v] + bias);
    Out[base + xa + 16] = (bf16)(acc1[v] + bias);
  }
}

// ---------------------------------------------------------------------------
// Depthwise 8x8 conv over the reflect-padded (H+1)x(W+1) field + BN (eval).
// 16x16 output tile per block; 23x23 input tile staged in LDS.
// ---------------------------------------------------------------------------
__global__ void __launch_bounds__(256)
k_dwbn(const bf16* __restrict__ A, const float* __restrict__ Wdw,
       const float* __restrict__ gamma, const float* __restrict__ beta,
       bf16* __restrict__ Out) {
  const int C = 256, H = 256, Wd = 256, HW = H * Wd;
  __shared__ bf16 sT[23][24];
  __shared__ float sW[64];
  const int tid = threadIdx.x;
  const int tile = blockIdx.x;
  const int tx0 = (tile & 15) << 4;
  const int ty0 = ((tile >> 4) & 15) << 4;
  const int c = (tile >> 8) & 255;
  const int b = tile >> 16;
  const bf16* Ap = A + (long)(b * C + c) * HW;

  if (tid < 64) sW[tid] = Wdw[c * 64 + tid];
#pragma unroll
  for (int e = tid; e < 23 * 23; e += 256) {
    int ry = e / 23, rx = e % 23;
    int t = ty0 + ry - 3;
    int u = tx0 + rx - 3;
    bf16 v = (bf16)0.f;
    if (t >= 0 && t <= H && u >= 0 && u <= Wd) {
      int tr = (t == H) ? H - 2 : t;
      int uc = (u == Wd) ? Wd - 2 : u;
      v = Ap[tr * Wd + uc];
    }
    sT[ry][rx] = v;
  }
  __syncthreads();

  const int lx = tid & 15, ly = tid >> 4;
  float acc = 0.f;
#pragma unroll
  for (int r = 0; r < 8; ++r)
#pragma unroll
    for (int s = 0; s < 8; ++s)
      acc += sW[r * 8 + s] * (float)sT[ly + r][lx + s];

  // BN eval: mean 0, var 1 -> gamma * x / sqrt(1+eps) + beta
  Out[(long)(b * C + c) * HW + (ty0 + ly) * Wd + tx0 + lx] =
      (bf16)(gamma[c] * acc * 0.9999950000374997f + beta[c]);
}

// ---------------------------------------------------------------------------
// host launcher
// ---------------------------------------------------------------------------
extern "C" void kernel_launch(void* const* d_in, const int* in_sizes, int n_in,
                              void* d_out, int out_size, void* d_ws,
                              size_t ws_size, hipStream_t stream) {
  (void)in_sizes; (void)n_in; (void)out_size; (void)ws_size;
  const float* x     = (const float*)d_in[0];
  const float* wqkv  = (const float*)d_in[1];
  const float* relT  = (const float*)d_in[2];
  const float* wx    = (const float*)d_in[3];
  const float* bx    = (const float*)d_in[4];
  const float* wy    = (const float*)d_in[5];
  const float* by    = (const float*)d_in[6];
  const float* wdw   = (const float*)d_in[7];
  const float* gamma = (const float*)d_in[8];
  const float* beta  = (const float*)d_in[9];
  const float* wpw   = (const float*)d_in[10];

  const long NX = 2L * 256 * 256 * 256;          // 33,554,432
  char* p = (char*)d_ws;
  auto carve = [&](size_t bytes) {
    char* r = p;
    p += (bytes + 255) & ~(size_t)255;
    return (void*)r;
  };
  bf16*  xbf   = (bf16*)carve(NX * 2);
  bf16*  wqkvb = (bf16*)carve(768L * 256 * 2);
  bf16*  wxb   = (bf16*)carve(256L * 2048 * 2);
  bf16*  wyb   = (bf16*)carve(256L * 2048 * 2);
  bf16*  wpwb  = (bf16*)carve(256L * 256 * 2);
  float* biasT = (float*)carve(16L * 64 * 64 * 4);
  bf16*  qkvb  = (bf16*)carve(3L * NX * 2);
  bf16*  obuf  = (bf16*)carve(NX * 2);
  bf16*  abuf  = (bf16*)carve(NX * 2);
  bf16*  bnbuf = (bf16*)carve(NX * 2);

  auto cvt = [&](const float* s, bf16* d, long n) {
    long n4 = n / 4;
    k_f32_to_bf16<<<(unsigned)((n4 + 255) / 256), 256, 0, stream>>>(s, d, n4);
  };
  cvt(x, xbf, NX);
  cvt(wqkv, wqkvb, 768L * 256);
  cvt(wx, wxb, 256L * 2048);        // [C][C*8] already row-major over K
  cvt(wy, wyb, 256L * 2048);
  cvt(wpw, wpwb, 256L * 256);
  k_bias_prep<<<256, 256, 0, stream>>>(relT, biasT);

  // qkv 1x1 conv: [768 x 256] @ [256 x 131072]
  {
    dim3 g(2048, 6);
    k_gemm_nchw<false><<<g, 256, 0, stream>>>(wqkvb, xbf, qkvb, nullptr,
                                              768, 256, 65536);
  }
  // window attention: 2048 windows x 16 heads
  k_attn<<<32768, 128, 0, stream>>>(qkvb, biasT, obuf);
  // fused strip convolutions (implicit GEMM, K = 2048 each)
  {
    dim3 g(2048, 4);
    k_strip<<<g, 256, 0, stream>>>(wxb, wyb, obuf, bx, by, abuf);
  }
  // depthwise 8x8 + BN
  k_dwbn<<<131072, 256, 0, stream>>>(abuf, wdw, gamma, beta, bnbuf);
  // pointwise projection: [256 x 256] @ [256 x 131072] -> fp32 out
  {
    dim3 g(2048, 2);
    k_gemm_nchw<true><<<g, 256, 0, stream>>>(wpwb, bnbuf, nullptr,
                                             (float*)d_out, 256, 256, 65536);
  }
}
